// MultiQueryAttention_53884659696210
// MI455X (gfx1250) — matrix operand order repaired and checked by reading, not compiled, not run on hardware
//
#include <hip/hip_runtime.h>

// ---- problem constants ----
#define D_MODEL 2048
#define N_HEADS 16
#define HEAD_DIM 128
#define BB 2
#define SS 2048
#define MTOT (BB * SS) // 4096

typedef _Float16 half8 __attribute__((ext_vector_type(8)));
typedef _Float16 v16h  __attribute__((ext_vector_type(16)));
typedef float    v8f   __attribute__((ext_vector_type(8)));

__device__ __forceinline__ v16h mk16(half8 lo, half8 hi) {
    return __builtin_shufflevector(lo, hi, 0, 1, 2, 3, 4, 5, 6, 7,
                                   8, 9, 10, 11, 12, 13, 14, 15);
}

// A-fragment (16x32, MxK): lane m<16 holds row m, K = hs*8..hs*8+7 then 16+hs*8..+7
__device__ __forceinline__ v16h load_afrag(const _Float16* base, int ld, int lane) {
    const int m = lane & 15, hs = lane >> 4;
    const _Float16* p = base + (size_t)m * ld + hs * 8;
    return mk16(*(const half8*)p, *(const half8*)(p + 16));
}

// B-fragment (32x16, KxN) for B[k][n] = Src[n][k] (Src row-major, K contiguous):
// lane n holds column n, K = hs*8..hs*8+7 then 16+hs*8..+7
__device__ __forceinline__ v16h load_bfrag(const _Float16* base, int ld, int lane) {
    const int n = lane & 15, hs = lane >> 4;
    const _Float16* p = base + (size_t)n * ld + hs * 8;
    return mk16(*(const half8*)p, *(const half8*)(p + 16));
}

__device__ __forceinline__ v8f wmma16(v16h a, v16h b, v8f c) {
    return __builtin_amdgcn_wmma_f32_16x16x32_f16(false, a, false, b,
                                                  (short)0, c, false, false);
}

// ---- CDNA5 async global->LDS copy (ASYNCcnt path) ----
// dsaddr is wave-relative; low 32 bits of a generic shared pointer are the
// AS(3) offset (generic->local lowering is a truncate).
__device__ __forceinline__ unsigned lds_off(const void* p) {
    return (unsigned)(size_t)p;
}
__device__ __forceinline__ void async_copy_b128(unsigned lds_addr, const _Float16* g) {
    asm volatile("global_load_async_to_lds_b128 %0, %1, off"
                 :: "v"(lds_addr), "v"((unsigned long long)(size_t)g)
                 : "memory");
}
__device__ __forceinline__ void wait_async0() {
#if __has_builtin(__builtin_amdgcn_s_wait_asynccnt)
    __builtin_amdgcn_s_wait_asynccnt(0);
#else
    asm volatile("s_wait_asynccnt 0x0" ::: "memory");
#endif
}

// ---------------------------------------------------------------------------
// fp32 -> f16 conversion
// ---------------------------------------------------------------------------
__global__ void cvt_f32_f16(const float* __restrict__ s, _Float16* __restrict__ d, int n) {
    int i = blockIdx.x * blockDim.x + threadIdx.x;
    if (i < n) d[i] = (_Float16)s[i];
}

// ---------------------------------------------------------------------------
// C(MxN) = A(MxK) * W(NxK)^T  -- async double-buffered LDS WMMA GEMM
// block = 256 threads (8 waves), tile 128x256, K-step 32
// wave (wy,wx): 64x64 sub-tile = 4x4 wmma accumulators (16 wmma / K-step)
// ---------------------------------------------------------------------------
template <typename OutT>
__global__ __launch_bounds__(256, 1) void gemm_xwt_wmma(
    const _Float16* __restrict__ A, const _Float16* __restrict__ W,
    OutT* __restrict__ C, int M, int N, int K) {
    __shared__ _Float16 As[2][128 * 32]; // 2 x 8 KB
    __shared__ _Float16 Ws[2][256 * 32]; // 2 x 16 KB

    const int tid  = threadIdx.x;
    const int lane = tid & 31;
    const int wave = tid >> 5;
    const int wy   = wave >> 2; // 0..1 -> 64-row slab
    const int wx   = wave & 3;  // 0..3 -> 64-col slab
    const int m0   = blockIdx.x * 128;
    const int n0   = blockIdx.y * 256;

    // cooperative async staging:
    //   As: thread t -> row t>>1, halves [(t&1)*16, +16)  (2 x b128)
    //   Ws: thread t -> row t,    halves [0, 32)          (4 x b128)
    const int ar = tid >> 1;
    const int ac = (tid & 1) * 16;
    const _Float16* ag = A + (size_t)(m0 + ar) * K + ac;
    const _Float16* wg = W + (size_t)(n0 + tid) * K;
    const unsigned asA0 = lds_off(&As[0][ar * 32 + ac]);
    const unsigned asA1 = lds_off(&As[1][ar * 32 + ac]);
    const unsigned asW0 = lds_off(&Ws[0][tid * 32]);
    const unsigned asW1 = lds_off(&Ws[1][tid * 32]);

    // prologue: stage K-tile 0 into buffer 0
    async_copy_b128(asA0,      ag);
    async_copy_b128(asA0 + 16, ag + 8);
    async_copy_b128(asW0,      wg);
    async_copy_b128(asW0 + 16, wg + 8);
    async_copy_b128(asW0 + 32, wg + 16);
    async_copy_b128(asW0 + 48, wg + 24);
    wait_async0();
    __syncthreads();

    v8f acc[4][4] = {};
    const int nk = K / 32;
    for (int kt = 0; kt < nk; ++kt) {
        const int cur = kt & 1;
        if (kt + 1 < nk) { // stream next K-tile while computing this one
            const _Float16* agn = ag + (size_t)(kt + 1) * 32;
            const _Float16* wgn = wg + (size_t)(kt + 1) * 32;
            const unsigned nA = cur ? asA0 : asA1;
            const unsigned nW = cur ? asW0 : asW1;
            async_copy_b128(nA,      agn);
            async_copy_b128(nA + 16, agn + 8);
            async_copy_b128(nW,      wgn);
            async_copy_b128(nW + 16, wgn + 8);
            async_copy_b128(nW + 32, wgn + 16);
            async_copy_b128(nW + 48, wgn + 24);
        }
        v16h bf[4];
#pragma unroll
        for (int nt = 0; nt < 4; ++nt)
            bf[nt] = load_bfrag(&Ws[cur][(wx * 64 + nt * 16) * 32], 32, lane);
#pragma unroll
        for (int mt = 0; mt < 4; ++mt) {
            v16h af = load_afrag(&As[cur][(wy * 64 + mt * 16) * 32], 32, lane);
#pragma unroll
            for (int nt = 0; nt < 4; ++nt)
                acc[mt][nt] = wmma16(af, bf[nt], acc[mt][nt]);
        }
        wait_async0();
        __syncthreads();
    }

    // epilogue: C layout -> element r is row (r + 8*(lane>=16)), col lane&15
    const int cn = lane & 15;
    const int rb = (lane >> 4) << 3;
#pragma unroll
    for (int mt = 0; mt < 4; ++mt) {
#pragma unroll
        for (int nt = 0; nt < 4; ++nt) {
            const size_t col  = (size_t)(n0 + wx * 64 + nt * 16 + cn);
            const size_t row0 = (size_t)(m0 + wy * 64 + mt * 16 + rb);
#pragma unroll
            for (int r = 0; r < 8; ++r)
                C[(row0 + r) * (size_t)N + col] = (OutT)acc[mt][nt][r];
        }
    }
}

// ---------------------------------------------------------------------------
// split KV (B*S x 256) into K (B,S,128) and V transposed (B,128,S)
// ---------------------------------------------------------------------------
__global__ void split_kv(const _Float16* __restrict__ KV, _Float16* __restrict__ Kd,
                         _Float16* __restrict__ Vt) {
    int i = blockIdx.x * blockDim.x + threadIdx.x;
    if (i >= BB * SS * HEAD_DIM) return;
    int d = i & (HEAD_DIM - 1);
    int s = (i >> 7) & (SS - 1);
    int b = i >> 18;
    size_t row = (size_t)b * SS + s;
    Kd[row * HEAD_DIM + d]                  = KV[row * 256 + d];
    Vt[((size_t)b * HEAD_DIM + d) * SS + s] = KV[row * 256 + 128 + d];
}

// ---------------------------------------------------------------------------
// Flash-attention: one wave per (b, h, 16-query tile). K/V shared across heads.
// Per 32-key block: 8 wmma for Q*K^T, online softmax (16-lane shfl reductions),
// P bounced via per-wave LDS buffer (C-layout -> A-fragment), 8 wmma for P*V.
// ---------------------------------------------------------------------------
__global__ __launch_bounds__(128, 2) void mqa_attention(
    const _Float16* __restrict__ Qh, const _Float16* __restrict__ Kh,
    const _Float16* __restrict__ Vt, _Float16* __restrict__ Ch) {
    __shared__ _Float16 pbuf[4][16 * 32];

    const int lane  = threadIdx.x & 31;
    const int wslot = threadIdx.x >> 5;
    const int gw    = blockIdx.x * 4 + wslot; // 0 .. B*H*(S/16)-1
    const int QT    = SS / 16;
    const int qt = gw % QT;
    const int h  = (gw / QT) % N_HEADS;
    const int b  = gw / (QT * N_HEADS);

    const int nl = lane & 15;  // column / row-in-frag index
    const int hs = lane >> 4;  // half select

    // Q fragments for this 16-row tile, full head dim (K = 128 -> 4 frags)
    v16h aQ[4];
    {
        const _Float16* qp = Qh + ((size_t)(b * SS + qt * 16 + nl)) * D_MODEL + h * HEAD_DIM;
#pragma unroll
        for (int ks = 0; ks < 4; ++ks) {
            const _Float16* p = qp + ks * 32 + hs * 8;
            aQ[ks] = mk16(*(const half8*)p, *(const half8*)(p + 16));
        }
    }

    v8f accO[8] = {}; // 16 x 128 output accumulator (8 d-tiles)
    float mrow[8], lrow[8];
#pragma unroll
    for (int r = 0; r < 8; ++r) { mrow[r] = -3.0e38f; lrow[r] = 0.0f; }

    const float scale = 0.08838834764831845f; // 1/sqrt(128)
    const int nkb = (qt * 16 + 15) / 32 + 1;  // causal: key blocks needed
    const _Float16* kbase = Kh + (size_t)b * SS * HEAD_DIM;
    const _Float16* vbase = Vt + (size_t)b * HEAD_DIM * SS;

    for (int kb = 0; kb < nkb; ++kb) {
        // ---- scores: Q(16x128) @ K_blk^T(128x32) ----
        v8f sc[2] = {};
#pragma unroll
        for (int nt = 0; nt < 2; ++nt) {
#pragma unroll
            for (int ks = 0; ks < 4; ++ks) {
                const _Float16* p =
                    kbase + (size_t)(kb * 32 + nt * 16 + nl) * HEAD_DIM + ks * 32 + hs * 8;
                v16h bk = mk16(*(const half8*)p, *(const half8*)(p + 16));
                sc[nt] = wmma16(aQ[ks], bk, sc[nt]);
            }
        }

        // ---- mask + scale + online softmax ----
        float mnew[8];
#pragma unroll
        for (int r = 0; r < 8; ++r) {
            const int rowg = qt * 16 + r + hs * 8;
#pragma unroll
            for (int nt = 0; nt < 2; ++nt) {
                const int colg = kb * 32 + nt * 16 + nl;
                float v = sc[nt][r] * scale;
                sc[nt][r] = (colg <= rowg) ? v : -3.0e38f;
            }
            float t = fmaxf(sc[0][r], sc[1][r]);
            t = fmaxf(t, __shfl_xor(t, 1, 32));
            t = fmaxf(t, __shfl_xor(t, 2, 32));
            t = fmaxf(t, __shfl_xor(t, 4, 32));
            t = fmaxf(t, __shfl_xor(t, 8, 32));
            mnew[r] = fmaxf(mrow[r], t);
        }
        float alpha[8];
#pragma unroll
        for (int r = 0; r < 8; ++r) {
            alpha[r] = __expf(mrow[r] - mnew[r]);
            float p0 = __expf(sc[0][r] - mnew[r]);
            float p1 = __expf(sc[1][r] - mnew[r]);
            sc[0][r] = p0;
            sc[1][r] = p1;
            float s = p0 + p1;
            s += __shfl_xor(s, 1, 32);
            s += __shfl_xor(s, 2, 32);
            s += __shfl_xor(s, 4, 32);
            s += __shfl_xor(s, 8, 32);
            lrow[r] = lrow[r] * alpha[r] + s;
            mrow[r] = mnew[r];
        }
#pragma unroll
        for (int dt = 0; dt < 8; ++dt)
#pragma unroll
            for (int r = 0; r < 8; ++r) accO[dt][r] *= alpha[r];

        // ---- redistribute P: C-layout -> LDS -> A-fragment layout ----
        _Float16* pb = pbuf[wslot];
#pragma unroll
        for (int nt = 0; nt < 2; ++nt)
#pragma unroll
            for (int r = 0; r < 8; ++r)
                pb[(r + hs * 8) * 32 + nt * 16 + nl] = (_Float16)sc[nt][r];
        // wave-internal DS ordering: compiler inserts s_wait_dscnt before reload
        v16h aP;
        {
            const _Float16* p = pb + nl * 32 + hs * 8;
            aP = mk16(*(const half8*)p, *(const half8*)(p + 16));
        }

        // ---- O += P(16x32) @ V_blk(32x128), V stored d-major (transposed) ----
#pragma unroll
        for (int dt = 0; dt < 8; ++dt) {
            const _Float16* p = vbase + (size_t)(dt * 16 + nl) * SS + kb * 32 + hs * 8;
            v16h bv = mk16(*(const half8*)p, *(const half8*)(p + 16));
            accO[dt] = wmma16(aP, bv, accO[dt]);
        }
    }

    // ---- normalize and write 16x128 tile into concat buffer ----
#pragma unroll
    for (int r = 0; r < 8; ++r) {
        const size_t rowg = (size_t)(b * SS + qt * 16 + r + hs * 8);
        const float inv = 1.0f / lrow[r];
#pragma unroll
        for (int dt = 0; dt < 8; ++dt)
            Ch[rowg * D_MODEL + h * HEAD_DIM + dt * 16 + nl] =
                (_Float16)(accO[dt][r] * inv);
    }
}

// ---------------------------------------------------------------------------
extern "C" void kernel_launch(void* const* d_in, const int* in_sizes, int n_in,
                              void* d_out, int out_size, void* d_ws, size_t ws_size,
                              hipStream_t stream) {
    (void)in_sizes; (void)n_in; (void)out_size; (void)ws_size;
    const float* X   = (const float*)d_in[0];
    const float* Wq  = (const float*)d_in[1];
    const float* Wkv = (const float*)d_in[2];
    const float* Wo  = (const float*)d_in[3];

    char* ws = (char*)d_ws;
    size_t off = 0;
    auto carve = [&](size_t elems) -> _Float16* {
        _Float16* p = (_Float16*)(ws + off);
        off += elems * sizeof(_Float16);
        off = (off + 255) & ~(size_t)255;
        return p;
    };
    _Float16* Xh   = carve((size_t)MTOT * D_MODEL);
    _Float16* Wqh  = carve((size_t)D_MODEL * D_MODEL);
    _Float16* Wkvh = carve((size_t)2 * HEAD_DIM * D_MODEL);
    _Float16* Woh  = carve((size_t)D_MODEL * D_MODEL);
    _Float16* Qh   = carve((size_t)MTOT * D_MODEL);
    _Float16* KVh  = carve((size_t)MTOT * 2 * HEAD_DIM);
    _Float16* Kh   = carve((size_t)MTOT * HEAD_DIM);
    _Float16* Vt   = carve((size_t)MTOT * HEAD_DIM);
    _Float16* Chh  = carve((size_t)MTOT * D_MODEL);

    // 1) fp32 -> f16 conversions
    {
        int n;
        n = MTOT * D_MODEL;
        cvt_f32_f16<<<(n + 255) / 256, 256, 0, stream>>>(X, Xh, n);
        n = D_MODEL * D_MODEL;
        cvt_f32_f16<<<(n + 255) / 256, 256, 0, stream>>>(Wq, Wqh, n);
        n = 2 * HEAD_DIM * D_MODEL;
        cvt_f32_f16<<<(n + 255) / 256, 256, 0, stream>>>(Wkv, Wkvh, n);
        n = D_MODEL * D_MODEL;
        cvt_f32_f16<<<(n + 255) / 256, 256, 0, stream>>>(Wo, Woh, n);
    }

    // 2) Q = X @ Wq^T  (4096 x 2048)
    gemm_xwt_wmma<_Float16><<<dim3(MTOT / 128, D_MODEL / 256), 256, 0, stream>>>(
        Xh, Wqh, Qh, MTOT, D_MODEL, D_MODEL);

    // 3) KV = X @ Wkv^T  (4096 x 256)
    gemm_xwt_wmma<_Float16><<<dim3(MTOT / 128, 1), 256, 0, stream>>>(
        Xh, Wkvh, KVh, MTOT, 256, D_MODEL);

    // 4) split K, transpose V
    {
        int n = BB * SS * HEAD_DIM;
        split_kv<<<(n + 255) / 256, 256, 0, stream>>>(KVh, Kh, Vt);
    }

    // 5) causal multi-query flash attention -> concat heads
    {
        int waves = BB * N_HEADS * (SS / 16); // 4096
        mqa_attention<<<waves / 4, 128, 0, stream>>>(Qh, Kh, Vt, Chh);
    }

    // 6) out = concat @ Wo^T  (fp32 output)
    gemm_xwt_wmma<float><<<dim3(MTOT / 128, D_MODEL / 256), 256, 0, stream>>>(
        Chh, Woh, (float*)d_out, MTOT, D_MODEL, D_MODEL);
}
